// MultiHeadAttention_35459249996475
// MI455X (gfx1250) — matrix operand order repaired and checked by reading, not compiled, not run on hardware
//
#include <hip/hip_runtime.h>
#include <hip/hip_bf16.h>

typedef __attribute__((ext_vector_type(16))) _Float16 v16h;
typedef __attribute__((ext_vector_type(8)))  _Float16 v8h;
typedef __attribute__((ext_vector_type(4)))  _Float16 v4h;
typedef __attribute__((ext_vector_type(8)))  float    v8f;
typedef __attribute__((ext_vector_type(4)))  float    v4f;

// Problem constants (fixed by the reference harness)
#define MHA_BT   32768     // 64*512
#define TILES    (MHA_BT / 2)          // 16384 two-item tiles (16 rows each)
#define WAVES_PB 4
// LDS: shared weight chunk (32KB) + per-wave {qh 8KB | kh 8KB | vt 8KB | wlds 256B}
#define WBUF_BYTES 32768
#define WAVE_LDS   24832
#define TOTAL_LDS  (WBUF_BYTES + WAVES_PB * WAVE_LDS)   // 132096 B -> 2 blocks/WGP

// ---------------------------------------------------------------------------
// Kernel 0: convert the three 256x256 f32 weight matrices to f16 in workspace
// ---------------------------------------------------------------------------
__global__ __launch_bounds__(256)
void cvt_weights_kernel(const float* __restrict__ wq,
                        const float* __restrict__ wk,
                        const float* __restrict__ wv,
                        _Float16* __restrict__ wh) {
  int i = blockIdx.x * blockDim.x + threadIdx.x;   // 0..65535
  wh[i]          = (_Float16)wq[i];
  wh[i + 65536]  = (_Float16)wk[i];
  wh[i + 131072] = (_Float16)wv[i];
}

// Build a 16-half fragment from two 8-half loads
__device__ __forceinline__ v16h cat16(v8h lo, v8h up) {
  v16h r;
  #pragma unroll
  for (int e = 0; e < 8; ++e) { r[e] = lo[e]; r[8 + e] = up[e]; }
  return r;
}

// ---------------------------------------------------------------------------
// Kernel 1: fused QKV projection + attention, both on WMMA.
// One wave owns 2 items (16 token-rows); the 4 waves of a block share each
// 32KB weight chunk staged in LDS (4x less L2 traffic than per-wave streams).
// ---------------------------------------------------------------------------
__global__ __launch_bounds__(128)
void mha_kernel(const float* __restrict__ emb,
                const _Float16* __restrict__ whp,
                const float* __restrict__ bq,
                const float* __restrict__ bk,
                const float* __restrict__ bv,
                float* __restrict__ out) {
  extern __shared__ char smem[];
  const int lane = threadIdx.x & 31;
  const int wave = threadIdx.x >> 5;
  const int tile = blockIdx.x * WAVES_PB + wave;   // 0..TILES-1

  _Float16* wbuf = (_Float16*)smem;                               // 64 W rows x 256
  _Float16* qh   = (_Float16*)(smem + WBUF_BYTES + wave * WAVE_LDS); // x staging -> Q
  _Float16* kh   = qh + 4096;                                     // 16x256 K row-major
  _Float16* vt   = qh + 8192;                                     // 256x16 V transposed
  _Float16* wlds = qh + 12288;                                    // 16x8 softmax weights

  // ---- stage x tile: 16 rows x 256 f32 -> f16 in LDS (coalesced b128 reads)
  {
    const v4f* src = (const v4f*)(emb + (size_t)tile * 4096);
    #pragma unroll
    for (int it = 0; it < 32; ++it) {
      int vi = it * 32 + lane;                 // 0..1023 float4's
      v4f f = src[vi];
      v4h hx = { (_Float16)f[0], (_Float16)f[1], (_Float16)f[2], (_Float16)f[3] };
      *(v4h*)(qh + vi * 4) = hx;
    }
  }

  const int row = lane & 15;   // A-row / C-column index
  const int hi  = lane >> 4;   // half-wave select

  // ---- load the 8 A fragments (16x32 f16) once; reused for Q,K,V ----
  // A layout: lanes 0-15 hold K = kb+{0..15},
  //           lanes 16-31 hold K = kb+{16..31}
  v16h afrag[8];
  #pragma unroll
  for (int kb = 0; kb < 8; ++kb) {
    const _Float16* xr = qh + row * 256 + kb * 32;
    afrag[kb] = cat16(*(const v8h*)(xr + hi * 16),
                      *(const v8h*)(xr + hi * 16 + 8));
  }

  // ---- projections: y[n,o] = sum_f x[n,f] * W[o,f] + b[o], via WMMA ----
  // Weights staged block-cooperatively in 32KB chunks (64 output rows each).
  const float* biases[3] = { bq, bk, bv };
  #pragma unroll
  for (int m = 0; m < 3; ++m) {
    const float* bias = biases[m];
    for (int ch = 0; ch < 4; ++ch) {
      __syncthreads();   // previous chunk fully consumed by all waves
      {
        const _Float16* wsrc = whp + (m << 16) + (ch << 14);
        #pragma unroll
        for (int cp = 0; cp < 32; ++cp) {
          int e = (cp * 128 + (int)threadIdx.x) * 8;   // v8h units -> halves
          *(v8h*)(wbuf + e) = *(const v8h*)(wsrc + e);
        }
      }
      __syncthreads();   // chunk visible to all waves

      #pragma unroll
      for (int otl = 0; otl < 4; ++otl) {
        const int ot = ch * 4 + otl;
        v8f acc = {};
        // B (32x16 = K x O): lane holds column o = ot*16 + row;
        // lanes 0-15 K=kb..kb+15, lanes 16-31 K=kb+16..kb+31 (contiguous halves)
        const _Float16* wp0 = wbuf + (otl * 16 + row) * 256 + hi * 16;
        #pragma unroll
        for (int kb = 0; kb < 8; ++kb) {
          v16h bfrag = *(const v16h*)(wp0 + kb * 32);
          acc = __builtin_amdgcn_wmma_f32_16x16x32_f16(
                    false, afrag[kb], false, bfrag, (short)0, acc, false, false);
        }
        const float bb = bias[ot * 16 + row];
        // C/D layout: VGPR j -> token row M = j + 8*hi; lane&15 -> column o
        #pragma unroll
        for (int j = 0; j < 8; ++j) {
          _Float16 hv = (_Float16)(acc[j] + bb);
          if (m == 2) vt[(ot * 16 + row) * 16 + (j + hi * 8)] = hv;  // V transposed
          else if (m == 1) kh[(j + hi * 8) * 256 + ot * 16 + row] = hv;
          else             qh[(j + hi * 8) * 256 + ot * 16 + row] = hv;
        }
      }
    }
  }

  // ---- attention via WMMA (per-wave, no barriers needed) ----
  // S^T = K * Q^T per head: each lane's 8 accumulators hold the 8 scores of one
  // query column c = lane&15; valid when hi == item(c). Softmax is in-lane.
  const int  myItem = row >> 3;                 // item of this lane's query column
  const bool valid  = (hi == myItem);
  const int  nloc   = row & 7;                  // query index within its item
  const bool amatch = (hi == (row >> 3));       // block-diag test for AV A-frag

  #pragma unroll
  for (int h = 0; h < 4; ++h) {
    // --- scores (transposed): 2 WMMAs over d = 64 ---
    v8f sacc = {};
    #pragma unroll
    for (int kb = 0; kb < 2; ++kb) {
      const _Float16* kr = kh + row * 256 + h * 64 + kb * 32;
      v16h af = cat16(*(const v8h*)(kr + hi * 16),
                      *(const v8h*)(kr + hi * 16 + 8));
      v16h bf = *(const v16h*)(qh + row * 256 + h * 64 + kb * 32 + hi * 16);
      sacc = __builtin_amdgcn_wmma_f32_16x16x32_f16(
                 false, af, false, bf, (short)0, sacc, false, false);
    }

    // --- in-lane softmax over the 8 scores (diagonal excluded -> weight 0) ---
    float s[8];
    #pragma unroll
    for (int j = 0; j < 8; ++j) s[j] = sacc[j] * 0.125f;   // 1/sqrt(64)
    float mx = -3.0e38f;
    #pragma unroll
    for (int j = 0; j < 8; ++j) if (j != nloc) mx = fmaxf(mx, s[j]);
    float sum = 0.f, w[8];
    #pragma unroll
    for (int j = 0; j < 8; ++j) {
      float e = (j == nloc) ? 0.f : __expf(s[j] - mx);
      w[j] = e; sum += e;
    }
    const float inv = 1.f / sum;
    v8h wv8;
    #pragma unroll
    for (int j = 0; j < 8; ++j) wv8[j] = (_Float16)(w[j] * inv);
    if (valid) *(v8h*)(wlds + row * 8) = wv8;   // same-wave DS ordering suffices

    // --- AV: attn[c][d] = sum_m w[c][m] * V[m][d], block-diagonal A ---
    // A-frag: row c; K slots 0..7 hold this item's 8 weights, all else zero pad
    v8h wrow = *(const v8h*)(wlds + row * 8);
    v16h wa;
    #pragma unroll
    for (int e = 0; e < 8; ++e) {
      wa[e]     = amatch ? wrow[e] : (_Float16)0;
      wa[8 + e] = (_Float16)0;
    }
    #pragma unroll
    for (int dc = 0; dc < 4; ++dc) {
      // B-frag: lanes 0-15 K=0..7, lanes 16-31 K=8..15 (m in vt); slots 8..15 zero pad
      v16h bv16;
      {
        v8h vlo = *(const v8h*)(vt + (h * 64 + dc * 16 + row) * 16 + hi * 8);
        #pragma unroll
        for (int e = 0; e < 8; ++e) { bv16[e] = vlo[e]; bv16[8 + e] = (_Float16)0; }
      }
      v8f oacc = {};
      oacc = __builtin_amdgcn_wmma_f32_16x16x32_f16(
                 false, wa, false, bv16, (short)0, oacc, false, false);
      // D: VGPR j -> query row j + 8*hi; lane&15 -> d column
      #pragma unroll
      for (int j = 0; j < 8; ++j) {
        out[((size_t)tile * 16 + j + 8 * hi) * 256 + h * 64 + dc * 16 + row] = oacc[j];
      }
    }
  }
}

// ---------------------------------------------------------------------------
// Launch
// ---------------------------------------------------------------------------
extern "C" void kernel_launch(void* const* d_in, const int* in_sizes, int n_in,
                              void* d_out, int out_size, void* d_ws, size_t ws_size,
                              hipStream_t stream) {
  const float* emb = (const float*)d_in[0];
  const float* Wq  = (const float*)d_in[1];
  const float* bq  = (const float*)d_in[2];
  const float* Wk  = (const float*)d_in[3];
  const float* bk  = (const float*)d_in[4];
  const float* Wv  = (const float*)d_in[5];
  const float* bv  = (const float*)d_in[6];
  float* outp = (float*)d_out;

  _Float16* whp = (_Float16*)d_ws;   // 3 * 65536 halves = 384 KB

  cvt_weights_kernel<<<256, 256, 0, stream>>>(Wq, Wk, Wv, whp);

  const int blocks = TILES / WAVES_PB;            // 4096
  mha_kernel<<<blocks, 128, TOTAL_LDS, stream>>>(emb, whp, bq, bk, bv, outp);
}
